// RotaryMHA_54185307407108
// MI455X (gfx1250) — matrix operand order repaired and checked by reading, not compiled, not run on hardware
//
#include <hip/hip_runtime.h>
#include <math.h>

// Problem constants (reference: B=2, T=2048, C=1024, H=16, D=64, ROT=32)
#define TSEQ 2048
#define CDIM 1024
#define NH   16
#define HD   64
#define MROW 4096   // B*T

typedef __bf16 bf16_t;
typedef __attribute__((ext_vector_type(16))) __bf16 bf16x16;
typedef __attribute__((ext_vector_type(8)))  float  f32x8;

union ABu { bf16x16 v; uint4 u[2]; };

// ---------------------------------------------------------------------------
// gfx1250 async LDS staging (GLOBAL_LOAD_ASYNC_TO_LDS_B128, ASYNCcnt-tracked).
// Builtin is typed as (v4i addrspace(1)*, v4i addrspace(3)*, imm offset, imm cpol)
// per the round-2 diagnostic. Guarded fallback: VGPR round-trip staging.
// ---------------------------------------------------------------------------
#if defined(__has_builtin)
#if __has_builtin(__builtin_amdgcn_global_load_async_to_lds_b128) && \
    __has_builtin(__builtin_amdgcn_s_wait_asynccnt)
#define USE_ASYNC_LDS 1
#endif
#endif

typedef int v4i_t __attribute__((vector_size(16)));
typedef __attribute__((address_space(1))) v4i_t* gas_v4i_p;
typedef __attribute__((address_space(3))) v4i_t* las_v4i_p;

__device__ __forceinline__ void stage16(void* lds_dst, const void* gsrc) {
#ifdef USE_ASYNC_LDS
    __builtin_amdgcn_global_load_async_to_lds_b128(
        (gas_v4i_p)const_cast<void*>(gsrc),
        (las_v4i_p)lds_dst,
        0, 0);
#else
    *(uint4*)lds_dst = *(const uint4*)gsrc;
#endif
}

__device__ __forceinline__ void stage_wait() {
#ifdef USE_ASYNC_LDS
    __builtin_amdgcn_s_wait_asynccnt(0);
#endif
}

__device__ __forceinline__ f32x8 zero8() {
    f32x8 z;
#pragma unroll
    for (int i = 0; i < 8; ++i) z[i] = 0.0f;
    return z;
}

// A-operand fragment, 16x32 bf16 (ISA 7.12.2): lane l -> M = l%16, kh = l/16.
// elements 0..7  : K = kh*8 + 0..7
// elements 8..15 : K = 16 + kh*8 + 0..7
__device__ __forceinline__ bf16x16 load_frag_A(const bf16_t* base, int lda, int lane) {
    int m  = lane & 15;
    int kh = lane >> 4;
    const bf16_t* row = base + m * lda;
    ABu r;
    r.u[0] = *(const uint4*)(row + kh * 8);
    r.u[1] = *(const uint4*)(row + 16 + kh * 8);
    return r.v;
}

// B-operand fragment, 32x16 bf16: lane l -> N = l%16, kh = l/16.
// elements 0..7  : K = kh*8 + 0..7
// elements 8..15 : K = 16 + kh*8 + 0..7
// Storage must be "transposed": rows indexed by N, contiguous in K.
__device__ __forceinline__ bf16x16 load_frag_B(const bf16_t* base, int ldb, int lane) {
    int n  = lane & 15;
    int kh = lane >> 4;
    const bf16_t* row = base + n * ldb;
    ABu r;
    r.u[0] = *(const uint4*)(row + kh * 8);
    r.u[1] = *(const uint4*)(row + 16 + kh * 8);
    return r.v;
}

__device__ __forceinline__ f32x8 wmma_bf16(bf16x16 a, bf16x16 b, f32x8 c) {
    return __builtin_amdgcn_wmma_f32_16x16x32_bf16(false, a, false, b, (short)0, c, false, false);
}

// ---------------------------------------------------------------------------
// Conversion kernels
// ---------------------------------------------------------------------------
__global__ __launch_bounds__(256) void cvt_bf16_kernel(const float* __restrict__ src,
                                                       bf16_t* __restrict__ dst, int n) {
    int i = blockIdx.x * 256 + threadIdx.x;
    if (i < n) dst[i] = (bf16_t)src[i];
}

// src[K=1024][N=1024] row-major -> dst[N][K] bf16 (pre-transpose weights so the
// WMMA B-operand is a contiguous-row LDS load)
__global__ __launch_bounds__(256) void cvt_transpose_kernel(const float* __restrict__ src,
                                                            bf16_t* __restrict__ dst) {
    int i = blockIdx.x * 256 + threadIdx.x;
    if (i < CDIM * CDIM) {
        int k = i / CDIM;
        int n = i % CDIM;
        dst[n * CDIM + k] = (bf16_t)src[i];
    }
}

// ---------------------------------------------------------------------------
// bf16 GEMM: Y[M,N] = X[M,1024] @ W + bias, BM=128 BN=64 BK=32, 8 waves.
// MODE 0: plain f32 store to out[M,1024]           (final Wo projection)
// MODE 1: RoPE epilogue, bf16 scatter to [B,H,T,D] (Q and K projections)
// MODE 2: bf16 scatter to [B,H,D,T] transposed     (V projection)
// ---------------------------------------------------------------------------
template <int MODE>
__global__ __launch_bounds__(256) void gemm_kernel(const bf16_t* __restrict__ X,
                                                   const bf16_t* __restrict__ Wt,
                                                   const float* __restrict__ bias,
                                                   void* __restrict__ outp) {
    constexpr int LDX = 40;  // 32 + 8 pad (keeps 16B alignment, spreads banks)
    constexpr int LDW = 40;
    __shared__ bf16_t Xs[128 * LDX];
    __shared__ bf16_t Ws[64 * LDW];

    const int tid  = threadIdx.x;
    const int lane = tid & 31;
    const int wave = tid >> 5;
    const int wm   = wave & 3;   // 4 row-strips of 32
    const int wn   = wave >> 2;  // 2 col-strips of 32
    const int m0   = blockIdx.y * 128;
    const int n0   = blockIdx.x * 64;

    f32x8 acc[2][2];
    acc[0][0] = zero8(); acc[0][1] = zero8();
    acc[1][0] = zero8(); acc[1][1] = zero8();

    for (int k0 = 0; k0 < CDIM; k0 += 32) {
        __syncthreads();
        // stage X tile 128x32 bf16 (512 x 16B, async to LDS)
#pragma unroll
        for (int t = 0; t < 2; ++t) {
            int j  = tid + t * 256;
            int r  = j >> 2;
            int cg = (j & 3) * 8;
            stage16(&Xs[r * LDX + cg], &X[(size_t)(m0 + r) * CDIM + k0 + cg]);
        }
        // stage W tile 64x32 bf16 (256 x 16B, async to LDS)
        {
            int r  = tid >> 2;
            int cg = (tid & 3) * 8;
            stage16(&Ws[r * LDW + cg], &Wt[(size_t)(n0 + r) * CDIM + k0 + cg]);
        }
        if (k0 + 32 < CDIM) {  // gfx1250 global_prefetch_b8 for the next K panel
            __builtin_prefetch(&X[(size_t)(m0 + (tid >> 1)) * CDIM + k0 + 32], 0, 1);
        }
        stage_wait();
        __syncthreads();

        bf16x16 a0 = load_frag_A(&Xs[(wm * 32 + 0) * LDX], LDX, lane);
        bf16x16 a1 = load_frag_A(&Xs[(wm * 32 + 16) * LDX], LDX, lane);
        bf16x16 b0 = load_frag_B(&Ws[(wn * 32 + 0) * LDW], LDW, lane);
        bf16x16 b1 = load_frag_B(&Ws[(wn * 32 + 16) * LDW], LDW, lane);

        acc[0][0] = wmma_bf16(a0, b0, acc[0][0]);
        acc[0][1] = wmma_bf16(a0, b1, acc[0][1]);
        acc[1][0] = wmma_bf16(a1, b0, acc[1][0]);
        acc[1][1] = wmma_bf16(a1, b1, acc[1][1]);
    }

    // Epilogue. C/D layout: col = lane&15, row = v + (lane>>4)*8.
    const int cl = lane & 15;
    const int rb = (lane >> 4) * 8;
    const float bia0 = bias[n0 + wn * 32 + cl];
    const float bia1 = bias[n0 + wn * 32 + 16 + cl];

#pragma unroll
    for (int mt = 0; mt < 2; ++mt) {
#pragma unroll
        for (int v = 0; v < 8; ++v) {
            int m = m0 + wm * 32 + mt * 16 + rb + v;
            int b = m >> 11;        // m / 2048
            int t = m & (TSEQ - 1); // m % 2048
            float a0 = acc[mt][0][v] + bia0;
            float a1 = acc[mt][1][v] + bia1;

            if (MODE == 0) {
                float* out = (float*)outp;
                out[(size_t)m * CDIM + n0 + wn * 32 + cl]      = a0;
                out[(size_t)m * CDIM + n0 + wn * 32 + 16 + cl] = a1;
            } else if (MODE == 1) {
                // RoPE on d in [0,32): pairs (d, d+16) are (ct0, ct1) at same lane.
                bf16_t* out = (bf16_t*)outp;
                int h = blockIdx.x;  // BN=64 == one head
                float r0 = a0, r1 = a1;
                if (wn == 0) {
                    // inv_freq[c] = 10000^(-c/16) = exp(-c * ln(1e4)/16)
                    float ang = (float)t * __expf(-0.5756462732485115f * (float)cl);
                    float sn, cs;
                    __sincosf(ang, &sn, &cs);
                    r0 = a0 * cs - a1 * sn;
                    r1 = a1 * cs + a0 * sn;
                }
                int d0 = wn * 32 + cl;
                size_t base = (((size_t)b * NH + h) * TSEQ + t) * HD;
                out[base + d0]      = (bf16_t)r0;
                out[base + d0 + 16] = (bf16_t)r1;
            } else {
                // V: store transposed [B,H,D,T] so it is a ready B-operand for P@V
                bf16_t* out = (bf16_t*)outp;
                int h  = blockIdx.x;
                int d0 = wn * 32 + cl;
                size_t base = ((size_t)b * NH + h) * HD * (size_t)TSEQ;
                out[base + (size_t)d0 * TSEQ + t]        = (bf16_t)a0;
                out[base + (size_t)(d0 + 16) * TSEQ + t] = (bf16_t)a1;
            }
        }
    }
}

// ---------------------------------------------------------------------------
// Flash attention, bf16 WMMA, online softmax.
// grid = (T/128 q-tiles, B*H). 256 threads = 8 waves; wave wq owns 16 query
// rows and the full D=64 output (4 WMMA column tiles).
// ---------------------------------------------------------------------------
__global__ __launch_bounds__(256) void attn_kernel(const bf16_t* __restrict__ Q,
                                                   const bf16_t* __restrict__ Kb,
                                                   const bf16_t* __restrict__ Vt,
                                                   bf16_t* __restrict__ Ob) {
    constexpr int LQ = 72, LK = 72, LV = 40, LP = 40;
    __shared__ bf16_t Qs[128 * LQ];        // 128 queries x 64 d
    __shared__ bf16_t Ks[32 * LK];         // 32 keys x 64 d  (B-operand for Q.K^T)
    __shared__ bf16_t Vs[64 * LV];         // 64 d x 32 keys  (B-operand for P.V)
    __shared__ bf16_t Ps[8 * 16 * LP];     // per-wave P patch: C-layout -> A-layout

    const int tid  = threadIdx.x;
    const int lane = tid & 31;
    const int wq   = tid >> 5;
    const int qt   = blockIdx.x;
    const int bh   = blockIdx.y;
    const int q0   = qt * 128;
    const size_t qkbase = (size_t)bh * TSEQ * HD;
    const size_t vbase  = (size_t)bh * HD * TSEQ;

    // stage Q tile once: 128x64 bf16 = 1024 x 16B (async to LDS)
#pragma unroll
    for (int t = 0; t < 4; ++t) {
        int j  = tid + t * 256;
        int r  = j >> 3;
        int cg = (j & 7) * 8;
        stage16(&Qs[r * LQ + cg], &Q[qkbase + (size_t)(q0 + r) * HD + cg]);
    }

    f32x8 o[4];
#pragma unroll
    for (int nt = 0; nt < 4; ++nt) o[nt] = zero8();
    float mrow[8], lrow[8];
#pragma unroll
    for (int v = 0; v < 8; ++v) { mrow[v] = -__builtin_inff(); lrow[v] = 0.0f; }

    const int cl = lane & 15;
    const int rb = (lane >> 4) * 8;
    const int nkt = qt * 4 + 4;  // causal: keys up to q0+127

    for (int kt = 0; kt < nkt; ++kt) {
        const int k0 = kt * 32;
        __syncthreads();
        {   // stage K tile: 32 keys x 64 d (256 x 16B)
            int r  = tid >> 3;
            int cg = (tid & 7) * 8;
            stage16(&Ks[r * LK + cg], &Kb[qkbase + (size_t)(k0 + r) * HD + cg]);
        }
        {   // stage Vt tile: 64 d x 32 keys (256 x 16B)
            int r  = tid >> 2;
            int cg = (tid & 3) * 8;
            stage16(&Vs[r * LV + cg], &Vt[vbase + (size_t)r * TSEQ + k0 + cg]);
        }
        stage_wait();
        __syncthreads();

        // S = Q . K^T  (two K=32 steps over d)
        f32x8 s0 = zero8(), s1 = zero8();
#pragma unroll
        for (int kk = 0; kk < 2; ++kk) {
            bf16x16 aq  = load_frag_A(&Qs[(wq * 16) * LQ + kk * 32], LQ, lane);
            bf16x16 bk0 = load_frag_B(&Ks[0 * LK + kk * 32], LK, lane);
            bf16x16 bk1 = load_frag_B(&Ks[16 * LK + kk * 32], LK, lane);
            s0 = wmma_bf16(aq, bk0, s0);
            s1 = wmma_bf16(aq, bk1, s1);
        }

        // scale + causal mask + online softmax (row reductions across 16 lanes)
#pragma unroll
        for (int v = 0; v < 8; ++v) {
            int qrow = q0 + wq * 16 + rb + v;
            float x0 = s0[v] * 0.125f;
            float x1 = s1[v] * 0.125f;
            if (k0 + cl > qrow)      x0 = -__builtin_inff();
            if (k0 + 16 + cl > qrow) x1 = -__builtin_inff();

            float mx = fmaxf(x0, x1);
#pragma unroll
            for (int off = 1; off < 16; off <<= 1) mx = fmaxf(mx, __shfl_xor(mx, off, 32));
            float mnew = fmaxf(mrow[v], mx);
            float corr = __expf(mrow[v] - mnew);
            float p0 = __expf(x0 - mnew);
            float p1 = __expf(x1 - mnew);
            float rs = p0 + p1;
#pragma unroll
            for (int off = 1; off < 16; off <<= 1) rs += __shfl_xor(rs, off, 32);
            lrow[v] = lrow[v] * corr + rs;
            mrow[v] = mnew;
#pragma unroll
            for (int nt = 0; nt < 4; ++nt) o[nt][v] *= corr;

            // P back to LDS in row-major (C-layout -> A-layout round trip)
            Ps[(wq * 16 + rb + v) * LP + cl]      = (bf16_t)p0;
            Ps[(wq * 16 + rb + v) * LP + 16 + cl] = (bf16_t)p1;
        }

        // O += P . V   (wave-private LDS patch; compiler inserts s_wait_dscnt)
        bf16x16 ap = load_frag_A(&Ps[(wq * 16) * LP], LP, lane);
#pragma unroll
        for (int nt = 0; nt < 4; ++nt) {
            bf16x16 bv = load_frag_B(&Vs[(nt * 16) * LV], LV, lane);
            o[nt] = wmma_bf16(ap, bv, o[nt]);
        }
    }

    // epilogue: normalize and write bf16 att output in [B,T,H*D] row-major
    const int b = bh >> 4;
    const int h = bh & 15;
#pragma unroll
    for (int v = 0; v < 8; ++v) {
        float inv = 1.0f / lrow[v];
        int t = q0 + wq * 16 + rb + v;
        size_t base = ((size_t)b * TSEQ + t) * CDIM + h * HD;
#pragma unroll
        for (int nt = 0; nt < 4; ++nt) {
            Ob[base + nt * 16 + cl] = (bf16_t)(o[nt][v] * inv);
        }
    }
}

// ---------------------------------------------------------------------------
extern "C" void kernel_launch(void* const* d_in, const int* in_sizes, int n_in,
                              void* d_out, int out_size, void* d_ws, size_t ws_size,
                              hipStream_t stream) {
    (void)in_sizes; (void)n_in; (void)out_size; (void)ws_size;
    const float* x  = (const float*)d_in[0];
    const float* Wq = (const float*)d_in[1];
    const float* bq = (const float*)d_in[2];
    const float* Wk = (const float*)d_in[3];
    const float* bk = (const float*)d_in[4];
    const float* Wv = (const float*)d_in[5];
    const float* bv = (const float*)d_in[6];
    const float* Wo = (const float*)d_in[7];
    const float* bo = (const float*)d_in[8];
    float* out = (float*)d_out;

    char* ws = (char*)d_ws;
    const size_t SZ_X = (size_t)MROW * CDIM * sizeof(bf16_t);  // 8 MB
    const size_t SZ_W = (size_t)CDIM * CDIM * sizeof(bf16_t);  // 2 MB
    bf16_t* xb   = (bf16_t*)(ws);
    bf16_t* wqt  = (bf16_t*)(ws + SZ_X);
    bf16_t* wkt  = (bf16_t*)(ws + SZ_X + SZ_W);
    bf16_t* wvt  = (bf16_t*)(ws + SZ_X + 2 * SZ_W);
    bf16_t* wot  = (bf16_t*)(ws + SZ_X + 3 * SZ_W);
    bf16_t* qb   = (bf16_t*)(ws + SZ_X + 4 * SZ_W);            // [B,H,T,D]
    bf16_t* kb   = (bf16_t*)(ws + 2 * SZ_X + 4 * SZ_W);        // [B,H,T,D]
    bf16_t* vtb  = (bf16_t*)(ws + 3 * SZ_X + 4 * SZ_W);        // [B,H,D,T]
    bf16_t* attb = (bf16_t*)(ws + 4 * SZ_X + 4 * SZ_W);        // [B*T, C]

    cvt_bf16_kernel<<<(MROW * CDIM) / 256, 256, 0, stream>>>(x, xb, MROW * CDIM);
    cvt_transpose_kernel<<<(CDIM * CDIM) / 256, 256, 0, stream>>>(Wq, wqt);
    cvt_transpose_kernel<<<(CDIM * CDIM) / 256, 256, 0, stream>>>(Wk, wkt);
    cvt_transpose_kernel<<<(CDIM * CDIM) / 256, 256, 0, stream>>>(Wv, wvt);
    cvt_transpose_kernel<<<(CDIM * CDIM) / 256, 256, 0, stream>>>(Wo, wot);

    dim3 gg(CDIM / 64, MROW / 128);  // (16, 32)
    gemm_kernel<1><<<gg, 256, 0, stream>>>(xb, wqt, bq, (void*)qb);
    gemm_kernel<1><<<gg, 256, 0, stream>>>(xb, wkt, bk, (void*)kb);
    gemm_kernel<2><<<gg, 256, 0, stream>>>(xb, wvt, bv, (void*)vtb);

    attn_kernel<<<dim3(TSEQ / 128, 2 * NH), 256, 0, stream>>>(qb, kb, vtb, attb);

    gemm_kernel<0><<<gg, 256, 0, stream>>>(attb, wot, bo, (void*)out);
}
